// KeyValueMask_11879879544515
// MI455X (gfx1250) — matrix operand order repaired and checked by reading, not compiled, not run on hardware
//
#include <hip/hip_runtime.h>
#include <math.h>

typedef __attribute__((ext_vector_type(16))) _Float16 v16h;
typedef __attribute__((ext_vector_type(8)))  _Float16 v8h;
typedef __attribute__((ext_vector_type(8)))  float    v8f;

#define NQ 8192
#define NK 8192
#define DK 128

#if __has_builtin(__builtin_amdgcn_exp2f)
#define EXP2F __builtin_amdgcn_exp2f
#else
#define EXP2F exp2f
#endif

// ---------------- pre-pass: K (f32 row-major) -> f16 row-major ----------------
__global__ void cvt_k_kernel(const float* __restrict__ K, _Float16* __restrict__ Kh) {
    int idx = blockIdx.x * blockDim.x + threadIdx.x;   // over NK*DK
    Kh[idx] = (_Float16)K[idx];
}

// ---------------- pre-pass: V (f32 [M][128]) -> f16 transposed [128][M] -------
__global__ void transpose_v_kernel(const float* __restrict__ V, _Float16* __restrict__ Vth) {
    __shared__ _Float16 tile[32][33];
    const int tx = threadIdx.x;            // 0..31
    const int ty = threadIdx.y;            // 0..7
    const int bx = blockIdx.x;             // 0..3   (cols of V)
    const int by = blockIdx.y;             // 0..255 (rows of V)
    #pragma unroll
    for (int i = ty; i < 32; i += 8)
        tile[i][tx] = (_Float16)V[(by * 32 + i) * DK + bx * 32 + tx];
    __syncthreads();
    #pragma unroll
    for (int i = ty; i < 32; i += 8)
        Vth[(bx * 32 + i) * NK + by * 32 + tx] = tile[tx][i];
}

// ---------------- main streaming-attention kernel ----------------------------
// block = 128 threads = 4 waves; each wave owns a 16-row Q tile over one key split.
// Scores are tiny (scale 1/128), so no max-shift is needed: accumulate
// unnormalized O and row-sums l; partials across splits are additive.
template <bool NORM>
__global__ __launch_bounds__(128)
void attn_fwd_kernel(const float* __restrict__ Q,
                     const _Float16* __restrict__ Kh,
                     const _Float16* __restrict__ Vth,
                     float* __restrict__ Opart,    // NORM: final output
                     float* __restrict__ Lpart,    // NORM: unused
                     int keys_per_split) {
    __shared__ _Float16 lds_p[4][16][40];   // per-wave P relayout scratch (padded)

    const int lane = threadIdx.x & 31;
    const int wv   = threadIdx.x >> 5;
    const int g    = lane >> 4;             // lane half (0/1)
    const int nx   = lane & 15;             // n for B/C fragments, m for A fragments
    const int qbase = blockIdx.x * 64 + wv * 16;
    const int split = blockIdx.y;
    const int klo = split * keys_per_split;
    const int khi = klo + keys_per_split;

    // (1/d_k)*log2(e) folded into Q so scores come out of WMMA in exp2 domain
    const float csc = 0.0078125f * 1.44269504088896340736f;

    // ---- load Q tile once, directly into WMMA A-fragment layout (4 x K32 chunks)
    v16h qa[4];
    const float* qrow = Q + (qbase + nx) * DK;
    #pragma unroll
    for (int kc = 0; kc < 4; ++kc) {
        #pragma unroll
        for (int h = 0; h < 16; ++h) {
            int d = 32 * kc + 16 * g + h;
            qa[kc][h] = (_Float16)(qrow[d] * csc);
        }
    }

    v8f o[8] = {};            // unnormalized O accumulators: 8 x (16x16) f32
    float lp[8];              // per-lane partial row sums
    #pragma unroll
    for (int r = 0; r < 8; ++r) lp[r] = 0.0f;

    for (int kb = klo; kb < khi; kb += 32) {
        if (kb + 32 < khi) {
            __builtin_prefetch(Kh + (kb + 32 + nx) * DK + 64 * g, 0, 0);
            __builtin_prefetch(Kh + (kb + 48 + nx) * DK + 64 * g, 0, 0);
        }

        // ---- issue ALL K-fragment loads first (one clause, overlapped waits)
        const _Float16* kp = Kh + (kb + nx) * DK + 16 * g;
        v16h kf[8];
        #pragma unroll
        for (int kc = 0; kc < 4; ++kc) {
            kf[kc]     = *(const v16h*)(kp + 32 * kc);            // keys kb..kb+15
            kf[kc + 4] = *(const v16h*)(kp + 16 * DK + 32 * kc);  // keys kb+16..kb+31
        }

        // ---- S = (Q*csc) K^T : two 16x16 column tiles, K=128 as 4 x K32 WMMAs
        v8f s0 = {}, s1 = {};
        #pragma unroll
        for (int kc = 0; kc < 4; ++kc) {
            s0 = __builtin_amdgcn_wmma_f32_16x16x32_f16(false, qa[kc], false, kf[kc],
                                                        (short)0, s0, false, false);
            s1 = __builtin_amdgcn_wmma_f32_16x16x32_f16(false, qa[kc], false, kf[kc + 4],
                                                        (short)0, s1, false, false);
        }

        // ---- issue ALL V-fragment loads now; latency hides behind exp2/relayout
        v16h vf[8];
        const _Float16* vp = Vth + nx * NK + kb + 16 * g;
        #pragma unroll
        for (int c = 0; c < 8; ++c)
            vf[c] = *(const v16h*)(vp + (size_t)(c * 16) * NK);

        // ---- P = exp2(S); accumulate per-lane row sums (no cross-lane work here)
        float e0[8], e1[8];
        #pragma unroll
        for (int r = 0; r < 8; ++r) {
            e0[r] = EXP2F(s0[r]);
            e1[r] = EXP2F(s1[r]);
            lp[r] += e0[r] + e1[r];
        }

        // ---- relayout P: C-fragment -> row-major f16 in LDS -> A-fragment
        #pragma unroll
        for (int r = 0; r < 8; ++r) {
            int row = r + 8 * g;
            lds_p[wv][row][nx]      = (_Float16)e0[r];
            lds_p[wv][row][nx + 16] = (_Float16)e1[r];
        }
        asm volatile("" ::: "memory");      // same-wave DS ops are in-order
        v8h plo = *(const v8h*)(&lds_p[wv][nx][16 * g]);
        v8h phi = *(const v8h*)(&lds_p[wv][nx][16 * g + 8]);
        v16h pa = __builtin_shufflevector(plo, phi, 0, 1, 2, 3, 4, 5, 6, 7,
                                          8, 9, 10, 11, 12, 13, 14, 15);

        // ---- O += P V : 8 output column tiles, K=32 (no rescale needed)
        #pragma unroll
        for (int c = 0; c < 8; ++c)
            o[c] = __builtin_amdgcn_wmma_f32_16x16x32_f16(false, pa, false, vf[c],
                                                          (short)0, o[c], false, false);
    }

    // ---- epilogue: reduce row sums across the 16-lane half, then write
    float lsum[8];
    #pragma unroll
    for (int r = 0; r < 8; ++r) {
        float rs = lp[r];
        rs += __shfl_xor(rs, 1, 32);
        rs += __shfl_xor(rs, 2, 32);
        rs += __shfl_xor(rs, 4, 32);
        rs += __shfl_xor(rs, 8, 32);
        lsum[r] = rs;
    }

    if (NORM) {
        #pragma unroll
        for (int c = 0; c < 8; ++c)
            #pragma unroll
            for (int r = 0; r < 8; ++r)
                Opart[(qbase + r + 8 * g) * DK + c * 16 + nx] = o[c][r] / lsum[r];
    } else {
        float* op = Opart + (size_t)split * NQ * DK;
        #pragma unroll
        for (int c = 0; c < 8; ++c)
            #pragma unroll
            for (int r = 0; r < 8; ++r)
                op[(qbase + r + 8 * g) * DK + c * 16 + nx] = o[c][r];
        if (nx == 0) {
            #pragma unroll
            for (int r = 0; r < 8; ++r)
                Lpart[split * NQ + qbase + r + 8 * g] = lsum[r];
        }
    }
}

// ---------------- combine: Out = (sum_s O_s) / (sum_s l_s) -------------------
__global__ void combine_kernel(const float* __restrict__ Opart,
                               const float* __restrict__ Lpart,
                               float* __restrict__ Out, int ksplit) {
    int idx = blockIdx.x * blockDim.x + threadIdx.x;   // over NQ*DK
    int row = idx >> 7;
    float acc = 0.0f, l = 0.0f;
    for (int s = 0; s < ksplit; ++s) {
        acc += Opart[(size_t)s * NQ * DK + idx];
        l   += Lpart[(size_t)s * NQ + row];
    }
    Out[idx] = acc / l;
}

extern "C" void kernel_launch(void* const* d_in, const int* in_sizes, int n_in,
                              void* d_out, int out_size, void* d_ws, size_t ws_size,
                              hipStream_t stream) {
    (void)in_sizes; (void)n_in; (void)out_size;
    const float* Q = (const float*)d_in[0];
    const float* K = (const float*)d_in[1];
    const float* V = (const float*)d_in[2];
    float* Out = (float*)d_out;

    _Float16* Kh  = (_Float16*)d_ws;              // 2 MB
    _Float16* Vth = Kh + (size_t)NK * DK;         // 2 MB (transposed V)
    const size_t base = (size_t)NK * DK * 2 * 2;  // 4 MB for Kh+Vth

    // pick largest key-split whose partials fit in workspace (deterministic)
    int ksplit = 4;
    while (ksplit > 1 &&
           base + (size_t)ksplit * ((size_t)NQ * DK + NQ) * sizeof(float) > ws_size)
        ksplit >>= 1;
    bool partial = base + (size_t)ksplit * ((size_t)NQ * DK + NQ) * sizeof(float) <= ws_size;

    cvt_k_kernel<<<(NK * DK) / 256, 256, 0, stream>>>(K, Kh);
    transpose_v_kernel<<<dim3(DK / 32, NK / 32), dim3(32, 8), 0, stream>>>(V, Vth);

    if (partial) {
        float* Opart = (float*)((char*)d_ws + base);
        float* Lpart = Opart + (size_t)ksplit * NQ * DK;
        attn_fwd_kernel<false><<<dim3(NQ / 64, ksplit), 128, 0, stream>>>(
            Q, Kh, Vth, Opart, Lpart, NK / ksplit);
        combine_kernel<<<(NQ * DK) / 256, 256, 0, stream>>>(Opart, Lpart, Out, ksplit);
    } else {
        attn_fwd_kernel<true><<<dim3(NQ / 64, 1), 128, 0, stream>>>(
            Q, Kh, Vth, Out, nullptr, NK);
    }
}
